// AutoCorrelation_11467562680395
// MI455X (gfx1250) — matrix operand order repaired and checked by reading, not compiled, not run on hardware
//
#include <hip/hip_runtime.h>
#include <hip/hip_bf16.h>
#include <math.h>
#include <stdint.h>

#define B_ 8
#define L_ 4096
#define D_ 512
#define KTOP 8   // int(1.0 * ln(4096)) == 8

typedef __attribute__((ext_vector_type(16))) _Float16 v16h;
typedef __attribute__((ext_vector_type(8)))  float    v8f;

union V16H { v16h v; float4 f4[2]; _Float16 h[16]; };
union H8   { float4 f4; _Float16 h[8]; };

// ------------------------------------------------ CDNA5 async-to-LDS helpers
__device__ __forceinline__ unsigned lds_addr_of(const void* p) {
  // generic -> LDS addrspace(3); its integer value is the LDS byte offset
  return (unsigned)(unsigned long long)(__attribute__((address_space(3))) const char*)p;
}

__device__ __forceinline__ void async_copy_b128(unsigned lds_off, const void* gptr) {
  // GLOBAL_LOAD_ASYNC_TO_LDS_B128, GV mode: vdst = per-lane LDS byte address,
  // vaddr = per-lane 64-bit global address, saddr = off. Tracked by ASYNCcnt.
  asm volatile("global_load_async_to_lds_b128 %0, %1, off"
               :: "v"(lds_off), "v"((unsigned long long)(uintptr_t)gptr)
               : "memory");
}

__device__ __forceinline__ void wait_async0() {
  asm volatile("s_wait_asynccnt 0" ::: "memory");
}

// ---------------------------------------------------------------- utilities
__global__ __launch_bounds__(256) void f32_to_f16_kernel(
    const float* __restrict__ x, _Float16* __restrict__ y, int n) {
  for (int i = blockIdx.x * 256 + threadIdx.x; i < n; i += gridDim.x * 256)
    y[i] = (_Float16)x[i];
}

__global__ __launch_bounds__(256) void zero_f32_kernel(float* __restrict__ p, int n) {
  int i = blockIdx.x * 256 + threadIdx.x;
  if (i < n) p[i] = 0.0f;
}

// ------------------------------------------------------------- WMMA GEMM
// C[M,N] = A[M,K] (row major, AT=float|f16) * Bw[K,N] (row major f16), out OT.
// Block = 256 threads = 8 waves; per-block tile 64(M) x 32(N); wave -> 16x16.
template <typename AT, typename OT>
__global__ __launch_bounds__(256) void gemm_wmma_kernel(
    const AT* __restrict__ A, const _Float16* __restrict__ Bw,
    OT* __restrict__ Cc, int M, int N, int K) {
  __shared__ _Float16 As[64][40];   // padded
  __shared__ _Float16 Bs[32][40];   // transposed: Bs[n][k], rows 16B aligned

  const int tid  = threadIdx.x;
  const int lane = tid & 31;        // wave32
  const int wv   = tid >> 5;        // 0..7
  const int wm   = wv >> 1;         // 0..3  (M sub-tile)
  const int wn   = wv & 1;          // 0..1  (N sub-tile)
  const int n    = lane & 15;       // A row m == B col n == lane&15
  const int hl   = lane >> 4;       // half-wave select

  const int rowBase = blockIdx.y * 64;
  const int cBase   = blockIdx.x * 32;

  v8f acc = {};

  for (int k0 = 0; k0 < K; k0 += 32) {
    // stage A tile 64x32 (convert to f16)
#pragma unroll
    for (int i = 0; i < 8; ++i) {
      int e = i * 256 + tid;
      int r = e >> 5, c = e & 31;
      As[r][c] = (_Float16)A[(size_t)(rowBase + r) * K + k0 + c];
    }
    // stage B tile 32x32 transposed
#pragma unroll
    for (int i = 0; i < 4; ++i) {
      int e = i * 256 + tid;
      int kr = e >> 5, c = e & 31;
      Bs[c][kr] = Bw[(size_t)(k0 + kr) * N + cBase + c];
    }
    __syncthreads();

    // A fragment (ISA 7.12.2): lane m=lane&15; VGPR j holds
    // K = (j<4?0:16) + 8*half + 2*(j&3), +1
    V16H a, b;
#pragma unroll
    for (int j = 0; j < 8; ++j) {
      int kk = ((j & 4) << 2) + (hl << 3) + ((j & 3) << 1);
      a.h[2 * j]     = As[16 * wm + n][kk];
      a.h[2 * j + 1] = As[16 * wm + n][kk + 1];
    }
    // B fragment: col n per lane, K = 8*half + 0..7, then 16 + 8*half + 0..7 -> 2x16B
    b.f4[0] = *reinterpret_cast<const float4*>(&Bs[16 * wn + n][hl << 3]);
    b.f4[1] = *reinterpret_cast<const float4*>(&Bs[16 * wn + n][(hl << 3) + 16]);

    acc = __builtin_amdgcn_wmma_f32_16x16x32_f16(false, a.v, false, b.v,
                                                 (short)0, acc, false, false);
    __syncthreads();
  }

#pragma unroll
  for (int r = 0; r < 8; ++r) {
    int row = rowBase + 16 * wm + r + 8 * hl;
    int col = cBase + 16 * wn + n;
    Cc[(size_t)row * N + col] = (OT)acc[r];
  }
}

// ---------------------------------------------- correlation (Gram diagonals)
// mean_corr[b,tau] += (1/D) * wrapped-diagonal sums of Q K^T.
// grid = (L/256, L/64, B). 8 waves/block; each wave owns 2 J (K time-blocks)
// and 4 I (Q time-blocks): every A fragment from LDS feeds 2 WMMAs and every
// B fragment from global feeds 4 WMMAs -> 8 wmma : 8 ds_b128 : 4 global_b128
// per k-step, balancing the LDS pipe against the matrix pipe.
__global__ __launch_bounds__(256) void corr_wmma_kernel(
    const _Float16* __restrict__ Qh, const _Float16* __restrict__ Kh,
    float* __restrict__ corr) {
  __shared__ _Float16 Qs[64 * 520];   // 64 rows x 512, padded stride 520 (~65KB)
  __shared__ float bins[8][248];      // per-wave bins: 2J x 4I x 31 diagonals

  const int tid  = threadIdx.x;
  const int lane = tid & 31;
  const int wv   = tid >> 5;
  const int n    = lane & 15;         // A row m == B col n
  const int hl   = lane >> 4;

  const int b  = blockIdx.z;
  const int I0 = blockIdx.y * 4;             // 4 Q time-blocks per workgroup
  const int J0 = (blockIdx.x * 8 + wv) * 2;  // 2 K time-blocks per wave

  // B operand rows: warm the caches while Qs stages (CDNA5 prefetch)
  const _Float16* krow0 = Kh + ((size_t)(b * L_ + J0 * 16 + n) << 9);
  const _Float16* krow1 = krow0 + ((size_t)16 << 9);
  __builtin_prefetch(krow0, 0, 0);
  __builtin_prefetch(krow0 + 512, 0, 0);
  __builtin_prefetch(krow1, 0, 0);
  __builtin_prefetch(krow1 + 512, 0, 0);

  // stage 64x512 Q strip via async DMA to LDS (no VGPR round-trip, ASYNCcnt)
#pragma unroll
  for (int i = 0; i < 16; ++i) {
    int cid = i * 256 + tid;              // 4096 x 16B chunks
    int r = cid >> 6, c8 = cid & 63;
    const void* src = Qh + ((size_t)(b * L_ + I0 * 16 + r) << 9) + (c8 << 3);
    async_copy_b128(lds_addr_of(&Qs[r * 520 + (c8 << 3)]), src);
  }
  for (int d = lane; d < 248; d += 32) bins[wv][d] = 0.0f;  // per-wave private
  wait_async0();
  __syncthreads();

  v8f acc[2][4] = {};
#pragma unroll
  for (int ks = 0; ks < 16; ++ks) {
    const int k0 = ks * 32;
    V16H b0, b1;
    b0.f4[0] = *reinterpret_cast<const float4*>(krow0 + k0 + (hl << 3));
    b0.f4[1] = *reinterpret_cast<const float4*>(krow0 + k0 + (hl << 3) + 16);
    b1.f4[0] = *reinterpret_cast<const float4*>(krow1 + k0 + (hl << 3));
    b1.f4[1] = *reinterpret_cast<const float4*>(krow1 + k0 + (hl << 3) + 16);
#pragma unroll
    for (int i = 0; i < 4; ++i) {
      V16H a;
      const int rowb = (i * 16 + n) * 520;
#pragma unroll
      for (int j = 0; j < 8; ++j) {
        int kk = k0 + ((j & 4) << 2) + (hl << 3) + ((j & 3) << 1);
        a.h[2 * j]     = Qs[rowb + kk];
        a.h[2 * j + 1] = Qs[rowb + kk + 1];
      }
      acc[0][i] = __builtin_amdgcn_wmma_f32_16x16x32_f16(false, a.v, false, b0.v,
                                                         (short)0, acc[0][i], false, false);
      acc[1][i] = __builtin_amdgcn_wmma_f32_16x16x32_f16(false, a.v, false, b1.v,
                                                         (short)0, acc[1][i], false, false);
    }
  }

  // reduce each 16x16 tile's 256 elements -> 31 diagonal bins (d = m - n + 15)
#pragma unroll
  for (int jj = 0; jj < 2; ++jj)
#pragma unroll
    for (int i = 0; i < 4; ++i)
#pragma unroll
      for (int r = 0; r < 8; ++r) {
        int mm = r + 8 * hl;
        atomicAdd(&bins[wv][(jj * 4 + i) * 31 + mm - n + 15], acc[jj][i][r]);
      }
  for (int d = lane; d < 248; d += 32) {
    int g = d / 31, dd = d - g * 31;
    int jj = g >> 2, i = g & 3;
    int tau = ((I0 + i - (J0 + jj)) * 16 + dd - 15) & (L_ - 1);
    atomicAdd(corr + ((size_t)b << 12) + tau, bins[wv][d] * (1.0f / 512.0f));
  }
}

// ------------------------------------------------------------ top-8 + softmax
__global__ __launch_bounds__(256) void topk_kernel(
    const float* __restrict__ corr, int* __restrict__ delays,
    float* __restrict__ weights) {
  const int b = blockIdx.x, tid = threadIdx.x;
  __shared__ float vals[L_];
  __shared__ float rv[256];
  __shared__ int   ri[256];
  __shared__ float topv[KTOP];
  __shared__ int   topi[KTOP];

  for (int i = tid; i < L_; i += 256) vals[i] = corr[(size_t)b * L_ + i];
  __syncthreads();

  for (int it = 0; it < KTOP; ++it) {
    float bv = -3.0e38f; int bi = 0;
    for (int i = tid; i < L_; i += 256) {
      float x = vals[i];
      if (x > bv) { bv = x; bi = i; }
    }
    rv[tid] = bv; ri[tid] = bi;
    __syncthreads();
    for (int s = 128; s > 0; s >>= 1) {
      if (tid < s && rv[tid + s] > rv[tid]) { rv[tid] = rv[tid + s]; ri[tid] = ri[tid + s]; }
      __syncthreads();
    }
    if (tid == 0) { topv[it] = rv[0]; topi[it] = ri[0]; vals[ri[0]] = -3.0e38f; }
    __syncthreads();
  }

  if (tid == 0) {
    float mx = topv[0];
    for (int i = 1; i < KTOP; ++i) mx = fmaxf(mx, topv[i]);
    float e[KTOP], s = 0.0f;
    for (int i = 0; i < KTOP; ++i) { e[i] = expf(topv[i] - mx); s += e[i]; }
    for (int i = 0; i < KTOP; ++i) {
      weights[b * KTOP + i] = e[i] / s;
      delays[b * KTOP + i]  = topi[i];
    }
  }
}

// ------------------------------------------- delay-gather weighted aggregation
__global__ __launch_bounds__(256) void aggregate_kernel(
    const _Float16* __restrict__ Vh, const int* __restrict__ delays,
    const float* __restrict__ weights, _Float16* __restrict__ agg) {
  const int g   = blockIdx.x * 256 + threadIdx.x;   // one 8-elem chunk each
  const int d8  = g & 63;
  const int row = g >> 6;
  const int t   = row & (L_ - 1);
  const int b   = row >> 12;

  float acc[8] = {0, 0, 0, 0, 0, 0, 0, 0};
#pragma unroll
  for (int k = 0; k < KTOP; ++k) {
    float w  = weights[b * KTOP + k];
    int  src = (t + delays[b * KTOP + k]) & (L_ - 1);
    H8 u;
    u.f4 = *reinterpret_cast<const float4*>(
        Vh + (((size_t)(b << 12) + src) << 9) + (d8 << 3));
#pragma unroll
    for (int i = 0; i < 8; ++i) acc[i] += w * (float)u.h[i];
  }
  H8 o;
#pragma unroll
  for (int i = 0; i < 8; ++i) o.h[i] = (_Float16)acc[i];
  *reinterpret_cast<float4*>(agg + ((size_t)row << 9) + (d8 << 3)) = o.f4;
}

// --------------------------------------------------------------------- launch
extern "C" void kernel_launch(void* const* d_in, const int* in_sizes, int n_in,
                              void* d_out, int out_size, void* d_ws, size_t ws_size,
                              hipStream_t stream) {
  (void)in_sizes; (void)n_in; (void)out_size; (void)ws_size;

  const float* q  = (const float*)d_in[0];
  const float* kx = (const float*)d_in[1];
  const float* vx = (const float*)d_in[2];
  const float* wq = (const float*)d_in[3];
  const float* wk = (const float*)d_in[4];
  const float* wv = (const float*)d_in[5];
  const float* wo = (const float*)d_in[6];

  const size_t DD  = (size_t)D_ * D_;        // 262,144
  const size_t BLD = (size_t)B_ * L_ * D_;   // 16,777,216
  const int    M   = B_ * L_;                // 32,768

  // workspace layout (~103 MB)
  _Float16* w16   = (_Float16*)d_ws;
  _Float16* wq16  = w16;
  _Float16* wk16  = w16 + DD;
  _Float16* wv16  = w16 + 2 * DD;
  _Float16* wo16  = w16 + 3 * DD;
  _Float16* Qh    = w16 + 4 * DD;
  _Float16* Kh    = Qh + BLD;
  _Float16* Vh    = Kh + BLD;
  float*    corr  = (float*)(Vh + BLD);
  int*      dly   = (int*)(corr + (size_t)B_ * L_);
  float*    wts   = (float*)(dly + B_ * KTOP);
  _Float16* agg16 = Qh;   // Qh dead after correlation -> reuse

  // 1) weights f32 -> f16
  f32_to_f16_kernel<<<64, 256, 0, stream>>>(wq, wq16, (int)DD);
  f32_to_f16_kernel<<<64, 256, 0, stream>>>(wk, wk16, (int)DD);
  f32_to_f16_kernel<<<64, 256, 0, stream>>>(wv, wv16, (int)DD);
  f32_to_f16_kernel<<<64, 256, 0, stream>>>(wo, wo16, (int)DD);

  // 2) projections (WMMA GEMMs, f32 A converted on stage-in)
  dim3 ggrid(D_ / 32, M / 64);
  gemm_wmma_kernel<float, _Float16><<<ggrid, 256, 0, stream>>>(q,  wq16, Qh, M, D_, D_);
  gemm_wmma_kernel<float, _Float16><<<ggrid, 256, 0, stream>>>(kx, wk16, Kh, M, D_, D_);
  gemm_wmma_kernel<float, _Float16><<<ggrid, 256, 0, stream>>>(vx, wv16, Vh, M, D_, D_);

  // 3) mean autocorrelation via Gram-tile diagonal sums
  zero_f32_kernel<<<(B_ * L_ + 255) / 256, 256, 0, stream>>>(corr, B_ * L_);
  corr_wmma_kernel<<<dim3(L_ / 256, L_ / 64, B_), 256, 0, stream>>>(Qh, Kh, corr);

  // 4) top-8 delays + softmax weights (per batch)
  topk_kernel<<<B_, 256, 0, stream>>>(corr, dly, wts);

  // 5) weighted circular-shift aggregation of V
  aggregate_kernel<<<(int)(BLD / 8 / 256), 256, 0, stream>>>(Vh, dly, wts, agg16);

  // 6) output projection -> f32 d_out
  gemm_wmma_kernel<_Float16, float><<<ggrid, 256, 0, stream>>>(agg16, wo16, (float*)d_out, M, D_, D_);
}
